// MultiHeadAttention_5695126634518
// MI455X (gfx1250) — matrix operand order repaired and checked by reading, not compiled, not run on hardware
//
#include <hip/hip_runtime.h>
#include <hip/hip_bf16.h>

// ---------------------------------------------------------------------------
// MultiHeadAttention (diag-attention variant) for MI455X / gfx1250, wave32.
// All GEMMs use V_WMMA_F32_16X16X32_BF16 (bf16 in, f32 accumulate).
// ---------------------------------------------------------------------------

#define Bc 4
#define Tc 2048
#define Cc 1024
#define Hc 16
#define HDc 64

typedef __attribute__((ext_vector_type(16))) __bf16 bf16x16;
typedef __attribute__((ext_vector_type(8)))  __bf16 bf16x8;
typedef __attribute__((ext_vector_type(8)))  float  f32x8;

union ABf { bf16x16 v; bf16x8 h[2]; };

__device__ __forceinline__ f32x8 wmma_bf16(bf16x16 a, bf16x16 b, f32x8 c) {
    // (neg_a, A, neg_b, B, c_mod, C, reuse_a, reuse_b)
    return __builtin_amdgcn_wmma_f32_16x16x32_bf16(false, a, false, b,
                                                   (short)0, c, false, false);
}

#define NEG_BIG (-1.0e30f)

// ---------------------------------------------------------------------------
// Elementwise f32 -> bf16
// ---------------------------------------------------------------------------
__global__ void cvt_bf16(const float* __restrict__ in, __bf16* __restrict__ out, int n) {
    int i = blockIdx.x * blockDim.x + threadIdx.x;
    if (i < n) out[i] = (__bf16)in[i];
}

// W[h][k][d] (f32) -> WT[h][d][k] (bf16): makes B-operand loads contiguous in K.
__global__ void cvt_wT(const float* __restrict__ in, __bf16* __restrict__ out) {
    int i = blockIdx.x * blockDim.x + threadIdx.x;     // over H*C*HD
    if (i >= Hc * Cc * HDc) return;
    int d = i % HDc;
    int k = (i / HDc) % Cc;
    int h = i / (Cc * HDc);
    out[((size_t)h * HDc + d) * Cc + k] = (__bf16)in[i];
}

// ---------------------------------------------------------------------------
// Q/K/V projection: [T x C] @ [C x HD] per (b,h), one 16x16 tile per wave.
// grid = (64, B*H, 3), block = 256 (8 waves). z selects Q/K/V.
// ---------------------------------------------------------------------------
__global__ __launch_bounds__(256) void qkv_gemm(
        const __bf16* __restrict__ Xbf,   // [B][T][C]
        const __bf16* __restrict__ WT,    // [3][H][HD][C]
        __bf16* __restrict__ QKV) {       // [3][B][H][T][HD]
    int lane = threadIdx.x & 31, wave = threadIdx.x >> 5;
    int tile = blockIdx.x * 8 + wave;     // 0..511
    int mt = tile >> 2;                   // T/16
    int nt = tile & 3;                    // HD/16
    int bh = blockIdx.y;
    int b = bh >> 4, h = bh & 15;
    int hi = lane >> 4, lo = lane & 15;

    const __bf16* Arow = Xbf + (size_t)b * Tc * Cc + (size_t)(mt * 16 + lo) * Cc;
    const __bf16* Brow = WT + (size_t)blockIdx.z * Hc * HDc * Cc
                            + ((size_t)h * HDc + nt * 16 + lo) * Cc;
    f32x8 acc = {};
    for (int kb = 0; kb < Cc; kb += 32) {
        ABf a;
        a.h[0] = *(const bf16x8*)(Arow + kb + 8 * hi);
        a.h[1] = *(const bf16x8*)(Arow + kb + 16 + 8 * hi);
        ABf bm;
        bm.h[0] = *(const bf16x8*)(Brow + kb + 8 * hi);
        bm.h[1] = *(const bf16x8*)(Brow + kb + 16 + 8 * hi);
        acc = wmma_bf16(a.v, bm.v, acc);
    }
    __bf16* Dst = QKV + (size_t)blockIdx.z * Bc * Hc * Tc * HDc
                      + ((size_t)bh * Tc + mt * 16) * HDc + nt * 16;
    for (int r = 0; r < 8; ++r) {
        int m = 8 * hi + r;
        Dst[(size_t)m * HDc + lo] = (__bf16)acc[r];
    }
}

// ---------------------------------------------------------------------------
// Fused logits + causal softmax. One workgroup = one 16-row strip of one
// (b,h) attention matrix. The 16 x 2048 logits panel lives in 128 KB LDS,
// so the 1 GB weights tensor is written exactly once.
// grid = (T/16, B*H), block = 256 (8 waves), dynamic LDS = 16*T*4 bytes.
// ---------------------------------------------------------------------------
__global__ __launch_bounds__(256) void logits_softmax(
        const __bf16* __restrict__ Qbf,   // [B*H][T][HD]
        const __bf16* __restrict__ Kbf,   // [B*H][T][HD]
        float* __restrict__ Wgt) {        // [B*H][T][T]
    extern __shared__ float smem[];       // [16][T]
    __shared__ float rowMax[16], rowInv[16];

    int strip = blockIdx.x;               // i-tile index
    int bh = blockIdx.y;
    int lane = threadIdx.x & 31, wave = threadIdx.x >> 5;
    int hi = lane >> 4, lo = lane & 15;
    int i0 = strip * 16;

    const __bf16* Qb = Qbf + (size_t)bh * Tc * HDc;
    const __bf16* Kb = Kbf + (size_t)bh * Tc * HDc;

    // A = 16 rows of K (reused across every j-tile of this wave)
    const __bf16* Arow = Kb + (size_t)(i0 + lo) * HDc;
    ABf a0, a1;
    a0.h[0] = *(const bf16x8*)(Arow + 0  + 8 * hi);
    a0.h[1] = *(const bf16x8*)(Arow + 16 + 8 * hi);
    a1.h[0] = *(const bf16x8*)(Arow + 32 + 8 * hi);
    a1.h[1] = *(const bf16x8*)(Arow + 48 + 8 * hi);

    // causal: only tiles jt <= strip carry live columns
    for (int jt = wave; jt <= strip; jt += 8) {
        const __bf16* Brow = Qb + (size_t)(jt * 16 + lo) * HDc;
        ABf b0, b1;
        b0.h[0] = *(const bf16x8*)(Brow + 0  + 8 * hi);
        b0.h[1] = *(const bf16x8*)(Brow + 16 + 8 * hi);
        b1.h[0] = *(const bf16x8*)(Brow + 32 + 8 * hi);
        b1.h[1] = *(const bf16x8*)(Brow + 48 + 8 * hi);
        f32x8 acc = {};
        acc = wmma_bf16(a0.v, b0.v, acc);     // d = 0..31
        acc = wmma_bf16(a1.v, b1.v, acc);     // d = 32..63
        int j0 = jt * 16;
        for (int r = 0; r < 8; ++r) {
            int m = 8 * hi + r;             // local row
            int gi = i0 + m, gj = j0 + lo;
            float v = acc[r] * 0.03125f;    // 1/sqrt(C) = 1/32
            smem[m * Tc + j0 + lo] = (gj <= gi) ? v : NEG_BIG;
        }
    }
    __syncthreads();

    int ncols = (strip + 1) * 16;           // live columns in this strip
    {   // per-row max & sum: 16 lanes per row, shuffle-reduce (wave32, width 16)
        int r = threadIdx.x >> 4;
        int c = threadIdx.x & 15;
        float mx = NEG_BIG;
        for (int j = c; j < ncols; j += 16) mx = fmaxf(mx, smem[r * Tc + j]);
        for (int off = 8; off >= 1; off >>= 1) mx = fmaxf(mx, __shfl_xor(mx, off, 16));
        float s = 0.f;
        for (int j = c; j < ncols; j += 16) s += __expf(smem[r * Tc + j] - mx);
        for (int off = 8; off >= 1; off >>= 1) s += __shfl_xor(s, off, 16);
        if (c == 0) { rowMax[r] = mx; rowInv[r] = 1.0f / s; }
    }
    __syncthreads();

    // coalesced write of the full 16 x T strip (zeros beyond the diagonal)
    float* Wout = Wgt + ((size_t)bh * Tc + i0) * (size_t)Tc;
    for (int idx = threadIdx.x; idx < 16 * Tc; idx += 256) {
        int r = idx >> 11;                  // T = 2048
        int j = idx & (Tc - 1);
        float val = 0.f;
        if (j < ncols)
            val = __expf(smem[r * Tc + j] - rowMax[r]) * rowInv[r];
        Wout[(size_t)r * Tc + j] = val;
    }
}

// ---------------------------------------------------------------------------
// cat[b][t][h*HD+d] = diag(weights)[b,h,t] * V[b,h,t,d]   (stored bf16 = A of
// the output-projection GEMM)
// ---------------------------------------------------------------------------
__global__ void scale_v(const __bf16* __restrict__ Vbf,
                        const float* __restrict__ Wgt,
                        __bf16* __restrict__ cat) {
    size_t idx = (size_t)blockIdx.x * blockDim.x + threadIdx.x;  // over B*T*C
    if (idx >= (size_t)Bc * Tc * Cc) return;
    int c = idx & (Cc - 1);
    size_t bt = idx >> 10;
    int t = (int)(bt & (Tc - 1));
    int b = (int)(bt >> 11);
    int h = c >> 6, d = c & 63;
    int bh = b * Hc + h;
    float diag = Wgt[((size_t)bh * Tc + t) * Tc + t];
    float v = (float)Vbf[((size_t)bh * Tc + t) * HDc + d];
    cat[idx] = (__bf16)(diag * v);
}

// ---------------------------------------------------------------------------
// out = cat @ Wp^T + bp : [B*T x C] @ [C x C]. One 16x16 tile per wave.
// grid = 4096, block = 256.
// ---------------------------------------------------------------------------
__global__ __launch_bounds__(256) void out_proj(
        const __bf16* __restrict__ cat,   // [B*T][C]
        const __bf16* __restrict__ Wpbf,  // [C(out)][C(in)]  (= B operand directly)
        const float* __restrict__ bp,
        float* __restrict__ out) {        // [B*T][C]
    int lane = threadIdx.x & 31, wave = threadIdx.x >> 5;
    int tile = blockIdx.x * 8 + wave;     // 0..32767
    int nt = tile & 63;                   // C/16
    int mt = tile >> 6;                   // B*T/16
    int hi = lane >> 4, lo = lane & 15;

    const __bf16* Arow = cat + (size_t)(mt * 16 + lo) * Cc;
    const __bf16* Brow = Wpbf + (size_t)(nt * 16 + lo) * Cc;
    f32x8 acc = {};
    for (int kb = 0; kb < Cc; kb += 32) {
        ABf a;
        a.h[0] = *(const bf16x8*)(Arow + kb + 8 * hi);
        a.h[1] = *(const bf16x8*)(Arow + kb + 16 + 8 * hi);
        ABf bm;
        bm.h[0] = *(const bf16x8*)(Brow + kb + 8 * hi);
        bm.h[1] = *(const bf16x8*)(Brow + kb + 16 + 8 * hi);
        acc = wmma_bf16(a.v, bm.v, acc);
    }
    int n = nt * 16 + lo;
    float bias = bp[n];
    for (int r = 0; r < 8; ++r) {
        int m = mt * 16 + 8 * hi + r;
        out[(size_t)m * Cc + n] = acc[r] + bias;
    }
}

// ---------------------------------------------------------------------------
// Workspace layout (bytes). Total ~92 MB.
// ---------------------------------------------------------------------------
#define XBF_OFF  ((size_t)0)                  // B*T*C bf16      = 16 MiB
#define WQT_OFF  ((size_t)16777216)           // 3 x H*HD*C bf16 =  6 MiB
#define WPB_OFF  ((size_t)23068672)           // C*C bf16        =  2 MiB
#define QKV_OFF  ((size_t)25165824)           // 3 x B*H*T*HD bf16 = 48 MiB
#define CAT_OFF  ((size_t)75497472)           // B*T*C bf16      = 16 MiB

extern "C" void kernel_launch(void* const* d_in, const int* in_sizes, int n_in,
                              void* d_out, int out_size, void* d_ws, size_t ws_size,
                              hipStream_t stream) {
    const float* x  = (const float*)d_in[0];
    const float* Wq = (const float*)d_in[1];
    const float* Wk = (const float*)d_in[2];
    const float* Wv = (const float*)d_in[3];
    const float* Wp = (const float*)d_in[4];
    const float* bp = (const float*)d_in[5];

    float* out = (float*)d_out;                       // [B][T][C]
    float* wgt = out + (size_t)Bc * Tc * Cc;          // [B][H][T][T]

    char* ws = (char*)d_ws;
    __bf16* Xbf  = (__bf16*)(ws + XBF_OFF);
    __bf16* WT   = (__bf16*)(ws + WQT_OFF);           // [3][H][HD][C]
    __bf16* Wpbf = (__bf16*)(ws + WPB_OFF);
    __bf16* QKV  = (__bf16*)(ws + QKV_OFF);           // [3][B][H][T][HD]
    __bf16* Qbf  = QKV;
    __bf16* Kbf  = QKV + (size_t)Bc * Hc * Tc * HDc;
    __bf16* Vbf  = QKV + (size_t)2 * Bc * Hc * Tc * HDc;
    __bf16* cat  = (__bf16*)(ws + CAT_OFF);

    // 1) precision conversion / weight transposes
    cvt_bf16<<<(Bc * Tc * Cc) / 256, 256, 0, stream>>>(x, Xbf, Bc * Tc * Cc);
    const float* wsrc[3] = {Wq, Wk, Wv};
    for (int m = 0; m < 3; ++m)
        cvt_wT<<<(Hc * Cc * HDc) / 256, 256, 0, stream>>>(
            wsrc[m], WT + (size_t)m * Hc * HDc * Cc);
    cvt_bf16<<<(Cc * Cc) / 256, 256, 0, stream>>>(Wp, Wpbf, Cc * Cc);

    // 2) Q/K/V projections (WMMA bf16)
    qkv_gemm<<<dim3(64, Bc * Hc, 3), 256, 0, stream>>>(Xbf, WT, QKV);

    // 3) fused logits + causal softmax -> weights (128 KB LDS per workgroup)
    (void)hipFuncSetAttribute((const void*)logits_softmax,
                              hipFuncAttributeMaxDynamicSharedMemorySize,
                              16 * Tc * (int)sizeof(float));
    logits_softmax<<<dim3(Tc / 16, Bc * Hc), 256, 16 * Tc * sizeof(float), stream>>>(
        Qbf, Kbf, wgt);

    // 4) diag-scaled V, concat heads
    scale_v<<<(Bc * Tc * Cc) / 256, 256, 0, stream>>>(Vbf, wgt, cat);

    // 5) output projection + bias (WMMA bf16)
    out_proj<<<(Bc * Tc / 16) * (Cc / 16) / 8, 256, 0, stream>>>(cat, Wpbf, bp, out);
}
